// GAT2_5875515261613
// MI455X (gfx1250) — compile-verified, matrix operand order checked by reading, not run on hardware
//
#include <hip/hip_runtime.h>
#include <math.h>

#define NEG_SLOPE 0.2f
#define N_GRAPHS 32

typedef __attribute__((ext_vector_type(16))) _Float16 v16h;
typedef __attribute__((ext_vector_type(8)))  _Float16 v8h;
typedef __attribute__((ext_vector_type(8)))  float    v8f;

// ---------------------------------------------------------------- utilities

__global__ __launch_bounds__(256) void fill_kernel(float* p, float v, long n) {
    long i = (long)blockIdx.x * blockDim.x + threadIdx.x;
    if (i < n) p[i] = v;
}

__device__ inline void atomicMaxFloat(float* addr, float val) {
    unsigned int* ua = (unsigned int*)addr;
    unsigned int old = *ua;
    while (__uint_as_float(old) < val) {
        unsigned int assumed = old;
        old = atomicCAS(ua, assumed, __float_as_uint(val));
        if (old == assumed) break;
    }
}

// f32 -> f16 row-major copy (GEMM A-operand staging)
__global__ __launch_bounds__(256) void cvt_f16_kernel(const float* __restrict__ in,
                                                      _Float16* __restrict__ out, long n) {
    long i = (long)blockIdx.x * blockDim.x + threadIdx.x;
    if (i < n) out[i] = (_Float16)in[i];
}

// W[K x N] f32 -> Wt[N x K] f16 (GEMM B-operand staging, transposed so each
// lane's fragment K-run is contiguous -> b128 loads)
__global__ __launch_bounds__(256) void transpose_cvt_f16_kernel(const float* __restrict__ W,
                                                                _Float16* __restrict__ Wt,
                                                                int K, int N) {
    long i = (long)blockIdx.x * blockDim.x + threadIdx.x;
    if (i >= (long)K * N) return;
    int n = (int)(i / K), k = (int)(i % K);
    Wt[(long)n * K + k] = (_Float16)W[(long)k * N + n];
}

// ------------------------------------------------- WMMA GEMM (f16 in, f32 acc)
// D[M x Ncol] = A[M x K] @ B[K x Ncol] (+bias, +relu).
// A: f16 row-major [M x K]. Bt: f16 [Ncol x K] (i.e. B transposed).
// Requires M % 16 == 0, Ncol % 128 == 0, K % 32 == 0 (K is a template constant
// so the k-loop fully unrolls into back-to-back v_wmma).
// Block = 256 threads = 8 waves; wave w owns column tile (blockIdx.y*8 + w).
template <int K>
__global__ __launch_bounds__(256) void wmma_gemm_kernel(const _Float16* __restrict__ A,
                                                        const _Float16* __restrict__ Bt,
                                                        const float* __restrict__ bias,
                                                        float* __restrict__ D,
                                                        int Ncol, int relu) {
    const int lane  = threadIdx.x & 31;          // full wave => EXEC all ones
    const int wave  = threadIdx.x >> 5;
    const int r     = lane & 15;
    const int kb    = (lane < 16) ? 0 : 8;       // fragment K-base per lane half
    const int tileM = blockIdx.x * 16;
    const int tileN = (blockIdx.y * 8 + wave) * 16;

    const _Float16* arow = A  + (long)(tileM + r) * K;   // lane's A row
    const _Float16* brow = Bt + (long)(tileN + r) * K;   // lane's B column

    v8f c = {};
#pragma unroll
    for (int k0 = 0; k0 < K; k0 += 32) {
        v8h alo = *(const v8h*)(arow + k0 + kb);
        v8h ahi = *(const v8h*)(arow + k0 + kb + 16);
        v8h blo = *(const v8h*)(brow + k0 + kb);
        v8h bhi = *(const v8h*)(brow + k0 + kb + 16);
        v16h a, b;
#pragma unroll
        for (int e = 0; e < 8; ++e) {
            a[e] = alo[e]; a[e + 8] = ahi[e];
            b[e] = blo[e]; b[e + 8] = bhi[e];
        }
        c = __builtin_amdgcn_wmma_f32_16x16x32_f16(false, a, false, b,
                                                   (short)0, c, false, false);
    }

    float bb = bias ? bias[tileN + r] : 0.0f;
#pragma unroll
    for (int v = 0; v < 8; ++v) {
        int row = tileM + v + ((lane < 16) ? 0 : 8);
        float x = c[v] + bb;
        if (relu) x = fmaxf(x, 0.0f);
        D[(long)row * Ncol + tileN + r] = x;
    }
}

// ---------------------------------------------- per-node attention scores el/er
__global__ __launch_bounds__(256) void scores_kernel(const float* __restrict__ feat,
                                                     const float* __restrict__ al,
                                                     const float* __restrict__ ar,
                                                     float* __restrict__ el,
                                                     float* __restrict__ er,
                                                     int N, int H, int D) {
    int wave = (blockIdx.x * blockDim.x + threadIdx.x) >> 5;
    int lane = threadIdx.x & 31;
    if (wave >= N) return;
    for (int h = 0; h < H; ++h) {
        float sl = 0.0f, sr = 0.0f;
        for (int d = lane; d < D; d += 32) {
            float f = feat[((long)wave * H + h) * D + d];
            sl += f * al[h * D + d];
            sr += f * ar[h * D + d];
        }
        for (int off = 16; off > 0; off >>= 1) {
            sl += __shfl_down(sl, off, 32);
            sr += __shfl_down(sr, off, 32);
        }
        if (lane == 0) { el[wave * H + h] = sl; er[wave * H + h] = sr; }
    }
}

// ------------------------------------------------------------- edge softmax
__global__ __launch_bounds__(256) void edge_max_kernel(const float* __restrict__ el,
                                                       const float* __restrict__ er,
                                                       const int* __restrict__ src,
                                                       const int* __restrict__ dst,
                                                       float* __restrict__ emax,
                                                       int E, int H) {
    int idx = blockIdx.x * blockDim.x + threadIdx.x;
    if (idx >= E * H) return;
    int e = idx / H, h = idx % H;
    float v = el[src[e] * H + h] + er[dst[e] * H + h];
    v = (v > 0.0f) ? v : NEG_SLOPE * v;
    atomicMaxFloat(&emax[dst[e] * H + h], v);
}

__global__ __launch_bounds__(256) void edge_sum_kernel(const float* __restrict__ el,
                                                       const float* __restrict__ er,
                                                       const int* __restrict__ src,
                                                       const int* __restrict__ dst,
                                                       const float* __restrict__ emax,
                                                       float* __restrict__ den,
                                                       int E, int H) {
    int idx = blockIdx.x * blockDim.x + threadIdx.x;
    if (idx >= E * H) return;
    int e = idx / H, h = idx % H;
    int t = dst[e];
    float v = el[src[e] * H + h] + er[t * H + h];
    v = (v > 0.0f) ? v : NEG_SLOPE * v;
    atomicAdd(&den[t * H + h], expf(v - emax[t * H + h]));
}

// one wave per (edge, head); lanes across D. rst[dst] += alpha * feat[src]
__global__ __launch_bounds__(256) void edge_aggregate_kernel(const float* __restrict__ el,
                                                             const float* __restrict__ er,
                                                             const float* __restrict__ emax,
                                                             const float* __restrict__ den,
                                                             const float* __restrict__ feat,
                                                             const int* __restrict__ src,
                                                             const int* __restrict__ dst,
                                                             float* __restrict__ rst,
                                                             int E, int H, int D) {
    int wid  = (blockIdx.x * blockDim.x + threadIdx.x) >> 5;
    int lane = threadIdx.x & 31;
    if (wid >= E * H) return;
    int e = wid / H, h = wid % H;
    int s = src[e], t = dst[e];
    const float* frow = &feat[((long)s * H + h) * D];
    __builtin_prefetch(frow, 0, 0);   // gfx1250 global_prefetch_b8
    float v = el[s * H + h] + er[t * H + h];
    v = (v > 0.0f) ? v : NEG_SLOPE * v;
    float ex = expf(v - emax[t * H + h]);
    float alpha = ex / fmaxf(den[t * H + h], 1e-9f);
    float* orow = &rst[((long)t * H + h) * D];
    for (int d = lane; d < D; d += 32)
        atomicAdd(&orow[d], alpha * frow[d]);
}

// out[n,f] = sum_h relu(rst[n,h,f] + bias[h,f])
__global__ __launch_bounds__(256) void head_sum_relu_kernel(const float* __restrict__ rst,
                                                            const float* __restrict__ bias,
                                                            float* __restrict__ out,
                                                            int N, int H, int D) {
    long idx = (long)blockIdx.x * blockDim.x + threadIdx.x;
    if (idx >= (long)N * D) return;
    int n = (int)(idx / D), f = (int)(idx % D);
    float acc = 0.0f;
    for (int h = 0; h < H; ++h)
        acc += fmaxf(rst[((long)n * H + h) * D + f] + bias[h * D + f], 0.0f);
    out[idx] = acc;
}

// ------------------------------------------------------------- graph pooling
__global__ __launch_bounds__(256) void pool_max_kernel(const float* __restrict__ x,
                                                       const int* __restrict__ gid,
                                                       float* __restrict__ pool,
                                                       int N, int D, int stride, int colOff) {
    long idx = (long)blockIdx.x * blockDim.x + threadIdx.x;
    if (idx >= (long)N * D) return;
    int n = (int)(idx / D), f = (int)(idx % D);
    atomicMaxFloat(&pool[gid[n] * stride + colOff + f], x[(long)n * D + f]);
}

__global__ __launch_bounds__(256) void pool_fixup_kernel(float* p, int n) {
    int i = blockIdx.x * blockDim.x + threadIdx.x;
    if (i < n && !isfinite(p[i])) p[i] = 0.0f;
}

// final: out[g] = relu(dot(mid[g,:], w) + b)
__global__ __launch_bounds__(32) void final_kernel(const float* __restrict__ mid,
                                                   const float* __restrict__ w,
                                                   const float* __restrict__ b,
                                                   float* __restrict__ out, int Kdim) {
    int g = blockIdx.x, lane = threadIdx.x;
    float acc = 0.0f;
    for (int k = lane; k < Kdim; k += 32) acc += mid[g * Kdim + k] * w[k];
    for (int off = 16; off > 0; off >>= 1) acc += __shfl_down(acc, off, 32);
    if (lane == 0) out[g] = fmaxf(acc + b[0], 0.0f);
}

// ---------------------------------------------------------------- host side

static inline int blks(long n, int b) { return (int)((n + b - 1) / b); }

static void run_branch(const float* x, const int* src, const int* dst, const int* gid,
                       const float* W1, const float* al1, const float* ar1, const float* b1,
                       const float* W2, const float* al2, const float* ar2, const float* b2,
                       int N, int E, int H1, int C1, int D1, int D2,
                       float* feat, float* rst, float* el, float* er,
                       float* emax, float* den, float* hbuf,
                       _Float16* ah, _Float16* w1t, _Float16* w2t,
                       float* pool, int colOff, hipStream_t stream) {
    // ---- layer 1: feat = x @ W1  (N x 64 @ 64 x 640), f16-staged WMMA
    cvt_f16_kernel<<<blks((long)N * D1, 256), 256, 0, stream>>>(x, ah, (long)N * D1);
    transpose_cvt_f16_kernel<<<blks((long)D1 * C1, 256), 256, 0, stream>>>(W1, w1t, D1, C1);
    wmma_gemm_kernel<64><<<dim3(N / 16, C1 / 128), 256, 0, stream>>>(
        ah, w1t, nullptr, feat, C1, 0);

    scores_kernel<<<blks((long)N * 32, 256), 256, 0, stream>>>(feat, al1, ar1, el, er, N, H1, D1);
    fill_kernel<<<blks((long)N * H1, 256), 256, 0, stream>>>(emax, -INFINITY, (long)N * H1);
    fill_kernel<<<blks((long)N * H1, 256), 256, 0, stream>>>(den, 0.0f, (long)N * H1);
    fill_kernel<<<blks((long)N * C1, 256), 256, 0, stream>>>(rst, 0.0f, (long)N * C1);
    edge_max_kernel<<<blks((long)E * H1, 256), 256, 0, stream>>>(el, er, src, dst, emax, E, H1);
    edge_sum_kernel<<<blks((long)E * H1, 256), 256, 0, stream>>>(el, er, src, dst, emax, den, E, H1);
    edge_aggregate_kernel<<<blks((long)E * H1 * 32, 256), 256, 0, stream>>>(
        el, er, emax, den, feat, src, dst, rst, E, H1, D1);
    head_sum_relu_kernel<<<blks((long)N * D1, 256), 256, 0, stream>>>(rst, b1, hbuf, N, H1, D1);

    // ---- layer 2: feat2 = h @ W2  (N x 64 @ 64 x 128); reuse feat/rst buffers
    cvt_f16_kernel<<<blks((long)N * D1, 256), 256, 0, stream>>>(hbuf, ah, (long)N * D1);
    transpose_cvt_f16_kernel<<<blks((long)D1 * D2, 256), 256, 0, stream>>>(W2, w2t, D1, D2);
    wmma_gemm_kernel<64><<<dim3(N / 16, D2 / 128), 256, 0, stream>>>(
        ah, w2t, nullptr, feat, D2, 0);

    scores_kernel<<<blks((long)N * 32, 256), 256, 0, stream>>>(feat, al2, ar2, el, er, N, 1, D2);
    fill_kernel<<<blks((long)N, 256), 256, 0, stream>>>(emax, -INFINITY, (long)N);
    fill_kernel<<<blks((long)N, 256), 256, 0, stream>>>(den, 0.0f, (long)N);
    fill_kernel<<<blks((long)N * D2, 256), 256, 0, stream>>>(rst, 0.0f, (long)N * D2);
    edge_max_kernel<<<blks((long)E, 256), 256, 0, stream>>>(el, er, src, dst, emax, E, 1);
    edge_sum_kernel<<<blks((long)E, 256), 256, 0, stream>>>(el, er, src, dst, emax, den, E, 1);
    edge_aggregate_kernel<<<blks((long)E * 32, 256), 256, 0, stream>>>(
        el, er, emax, den, feat, src, dst, rst, E, 1, D2);
    // out2 = relu(rst2 + b2) written over feat (free now)
    head_sum_relu_kernel<<<blks((long)N * D2, 256), 256, 0, stream>>>(rst, b2, feat, N, 1, D2);

    // ---- graph max-pool into pooled[32 x 256] at column offset
    pool_max_kernel<<<blks((long)N * D2, 256), 256, 0, stream>>>(
        feat, gid, pool, N, D2, 2 * D2, colOff);
}

extern "C" void kernel_launch(void* const* d_in, const int* in_sizes, int n_in,
                              void* d_out, int out_size, void* d_ws, size_t ws_size,
                              hipStream_t stream) {
    (void)n_in; (void)out_size; (void)ws_size;
    const float* lig_x   = (const float*)d_in[0];
    const int*   lig_src = (const int*)d_in[1];
    const int*   lig_dst = (const int*)d_in[2];
    const int*   lig_gid = (const int*)d_in[3];
    const float* rec_x   = (const float*)d_in[4];
    const int*   rec_src = (const int*)d_in[5];
    const int*   rec_dst = (const int*)d_in[6];
    const int*   rec_gid = (const int*)d_in[7];
    const float *W1l = (const float*)d_in[8],  *al1l = (const float*)d_in[9];
    const float *ar1l = (const float*)d_in[10], *b1l = (const float*)d_in[11];
    const float *W2l = (const float*)d_in[12], *al2l = (const float*)d_in[13];
    const float *ar2l = (const float*)d_in[14], *b2l = (const float*)d_in[15];
    const float *W1r = (const float*)d_in[16], *al1r = (const float*)d_in[17];
    const float *ar1r = (const float*)d_in[18], *b1r = (const float*)d_in[19];
    const float *W2r = (const float*)d_in[20], *al2r = (const float*)d_in[21];
    const float *ar2r = (const float*)d_in[22], *b2r = (const float*)d_in[23];
    const float *Wlin1 = (const float*)d_in[24], *blin1 = (const float*)d_in[25];
    const float *Wlin2 = (const float*)d_in[26], *blin2 = (const float*)d_in[27];
    float* out = (float*)d_out;

    const int D1 = 64;
    const int Nl = in_sizes[0] / D1, El = in_sizes[1];
    const int Nr = in_sizes[4] / D1, Er = in_sizes[5];
    const int C1 = in_sizes[8] / D1;     // 640
    const int H1 = in_sizes[9] / D1;     // 10
    const int D2 = in_sizes[12] / D1;    // 128
    const int Nmax = (Nl > Nr) ? Nl : Nr;

    // carve workspace (buffers sized for the larger branch, reused by both)
    char* ws = (char*)d_ws;
    size_t off = 0;
    auto carve = [&](size_t bytes) {
        void* p = ws + off;
        off += (bytes + 255) & ~(size_t)255;
        return p;
    };
    float* feat = (float*)carve((size_t)Nmax * C1 * 4);
    float* rst  = (float*)carve((size_t)Nmax * C1 * 4);
    float* el   = (float*)carve((size_t)Nmax * H1 * 4);
    float* er   = (float*)carve((size_t)Nmax * H1 * 4);
    float* emax = (float*)carve((size_t)Nmax * H1 * 4);
    float* den  = (float*)carve((size_t)Nmax * H1 * 4);
    float* hbuf = (float*)carve((size_t)Nmax * D1 * 4);
    float* pool = (float*)carve((size_t)N_GRAPHS * 2 * D2 * 4);
    float* mid  = (float*)carve((size_t)N_GRAPHS * D2 * 4);
    _Float16* ah    = (_Float16*)carve((size_t)Nmax * D1 * 2);       // GEMM A staging
    _Float16* w1t   = (_Float16*)carve((size_t)D1 * C1 * 2);         // W1^T f16
    _Float16* w2t   = (_Float16*)carve((size_t)D1 * D2 * 2);         // W2^T f16
    _Float16* poolh = (_Float16*)carve((size_t)N_GRAPHS * 2 * D2 * 2);
    _Float16* wl1t  = (_Float16*)carve((size_t)2 * D2 * D2 * 2);     // Wlin1^T f16

    // pooled features init to -inf (segment_max identity)
    fill_kernel<<<blks(N_GRAPHS * 2 * D2, 256), 256, 0, stream>>>(
        pool, -INFINITY, (long)N_GRAPHS * 2 * D2);

    run_branch(lig_x, lig_src, lig_dst, lig_gid, W1l, al1l, ar1l, b1l,
               W2l, al2l, ar2l, b2l, Nl, El, H1, C1, D1, D2,
               feat, rst, el, er, emax, den, hbuf, ah, w1t, w2t, pool, 0, stream);
    run_branch(rec_x, rec_src, rec_dst, rec_gid, W1r, al1r, ar1r, b1r,
               W2r, al2r, ar2r, b2r, Nr, Er, H1, C1, D1, D2,
               feat, rst, el, er, emax, den, hbuf, ah, w1t, w2t, pool, D2, stream);

    pool_fixup_kernel<<<blks(N_GRAPHS * 2 * D2, 256), 256, 0, stream>>>(
        pool, N_GRAPHS * 2 * D2);

    // MLP head: mid = relu(pool[32x256] @ Wlin1[256x128] + blin1)  (WMMA, K=256)
    cvt_f16_kernel<<<blks((long)N_GRAPHS * 2 * D2, 256), 256, 0, stream>>>(
        pool, poolh, (long)N_GRAPHS * 2 * D2);
    transpose_cvt_f16_kernel<<<blks((long)2 * D2 * D2, 256), 256, 0, stream>>>(
        Wlin1, wl1t, 2 * D2, D2);
    wmma_gemm_kernel<256><<<dim3(N_GRAPHS / 16, D2 / 128), 256, 0, stream>>>(
        poolh, wl1t, blin1, mid, D2, 1);

    // out = relu(mid @ Wlin2[128x1] + blin2)
    final_kernel<<<N_GRAPHS, 32, 0, stream>>>(mid, Wlin2, blin2, out, D2);
}
